// RAGSequentialRec_44092134261038
// MI455X (gfx1250) — compile-verified, matrix operand order checked by reading, not run on hardware
//
#include <hip/hip_runtime.h>

#define B_DIM 2048
#define K_RET 50
#define V_DIM 100000
#define H_DIM 512

typedef __bf16 bf16_t;
typedef __attribute__((ext_vector_type(16))) __bf16 v16bf;
typedef __attribute__((ext_vector_type(8)))  __bf16 v8bf;
typedef __attribute__((ext_vector_type(2)))  __bf16 v2bf;
typedef __attribute__((ext_vector_type(8)))  float  v8f;

union FragU { v16bf v; v8bf h[2]; };

typedef __attribute__((address_space(3))) void lds_void;

// LDS byte offset of a generic pointer that points into LDS.
__device__ __forceinline__ uint32_t lds_addr_of(void* p) {
  return (uint32_t)(uintptr_t)(lds_void*)p;
}

// CDNA5 async DMA copy: 16B global -> LDS per lane, tracked by ASYNCcnt.
__device__ __forceinline__ void async_copy_b128(uint32_t lds_byte_addr, const void* gptr) {
  asm volatile("global_load_async_to_lds_b128 %0, %1, off"
               :: "v"(lds_byte_addr), "v"(gptr) : "memory");
}

__device__ __forceinline__ void wait_asynccnt0() {
  asm volatile("s_wait_asynccnt 0x0" ::: "memory");
}

// CDNA5 16-bit A/B operand layout: lanes 0-15 hold K={kb..kb+7, kb+16..kb+23}
// with kb=0; lanes 16-31 the same with kb=8.  Two ds_load_b128 per lane.
__device__ __forceinline__ v16bf load_frag_row(const bf16_t* rowp, int kbase) {
  FragU f;
  f.h[0] = *(const v8bf*)(rowp + kbase);
  f.h[1] = *(const v8bf*)(rowp + kbase + 16);
  return f.v;
}

// ---------------------------------------------------------------------------
// Kernel 1: gather + mean-pool + concat + f32->bf16 convert.
// ---------------------------------------------------------------------------
__global__ __launch_bounds__(256)
void rag_prep_kernel(const float* __restrict__ user_rep,
                     const int*   __restrict__ indices,
                     const float* __restrict__ item_emb,
                     bf16_t*      __restrict__ A_fuse) {
  __shared__ int sidx[K_RET];
  const int b = blockIdx.y;
  const int h = blockIdx.x * 256 + threadIdx.x;
  if (threadIdx.x < K_RET) sidx[threadIdx.x] = indices[b * K_RET + threadIdx.x];
  __syncthreads();
  float s = 0.f;
  #pragma unroll 5
  for (int k = 0; k < K_RET; ++k)
    s += item_emb[(size_t)sidx[k] * H_DIM + h];
  A_fuse[(size_t)b * (2 * H_DIM) + h]         = (bf16_t)user_rep[(size_t)b * H_DIM + h];
  A_fuse[(size_t)b * (2 * H_DIM) + H_DIM + h] = (bf16_t)(s * (1.0f / (float)K_RET));
}

// ---------------------------------------------------------------------------
// Tiled WMMA GEMM: out[M][N] = A[M][Kd](bf16) * Bf[Kd][N](f32 -> bf16 on the
// fly) + bias.  Block tile 128x128x32, 8 waves, wave tile 64x32 (4x2 WMMA).
// A tile: async global->LDS DMA (no VGPRs).  B tile: f32 register prefetch,
// pack to bf16 pairs, ds_store_b32.  Double-buffered LDS, one barrier/K-step.
// blockIdx.x walks M-tiles (fast) so each B tile stays hot in L2 across all
// 16 M-tiles of one N-tile column.
// ---------------------------------------------------------------------------
template <bool OUT_BF16>
__global__ __launch_bounds__(256, 2)
void gemm_bf16_tn(const bf16_t* __restrict__ A,    // [M][Kd] row-major
                  const float*  __restrict__ Bf,   // [Kd][N] row-major f32
                  const float*  __restrict__ bias, // [N]
                  void* __restrict__ outp,         // [M][N] bf16 or f32
                  int M, int N, int Kd) {
  __shared__ bf16_t As[2][128][40];  // 80B row stride, 16B aligned chunks
  __shared__ bf16_t Bs[2][128][40];  // transposed: Bs[.][n][k]

  const int t    = threadIdx.x;
  const int lane = t & 31;
  const int wave = t >> 5;
  const int wmb  = (wave >> 2) * 64;
  const int wnb  = (wave & 3) * 32;
  const int m0   = blockIdx.x * 128;
  const int n0   = blockIdx.y * 128;
  const int lrow  = lane & 15;
  const int kbase = (lane < 16) ? 0 : 8;

  // A tile staging: async DMA, item i = t+256*it: r=i>>2, c=(i&3)*8 (16B chunk)
  auto stage_A_async = [&](int buf, int k0) {
    #pragma unroll
    for (int it = 0; it < 2; ++it) {
      int i = t + it * 256;
      int r = i >> 2, c = (i & 3) * 8;
      async_copy_b128(lds_addr_of(&As[buf][r][c]),
                      (const void*)(A + (size_t)(m0 + r) * Kd + k0 + c));
    }
  };

  // B tile: item i = t+256*it: k-pair p=i>>5, n=(i&31)*4; branch-free edge.
  float4 b0pre[2], b1pre[2];
  auto load_B = [&](int k0) {
    #pragma unroll
    for (int it = 0; it < 2; ++it) {
      int i = t + it * 256;
      int p = i >> 5, n = (i & 31) * 4;
      bool ok = (n0 + n) < N;              // N % 4 == 0
      int  ncl = ok ? (n0 + n) : 0;
      const float* p0 = Bf + (size_t)(k0 + 2 * p) * N + ncl;
      float4 f0 = *(const float4*)p0;
      float4 f1 = *(const float4*)(p0 + N);
      if (!ok) {
        f0 = make_float4(0.f, 0.f, 0.f, 0.f);
        f1 = make_float4(0.f, 0.f, 0.f, 0.f);
      }
      b0pre[it] = f0;
      b1pre[it] = f1;
    }
  };
  auto store_B = [&](int buf) {
    #pragma unroll
    for (int it = 0; it < 2; ++it) {
      int i = t + it * 256;
      int p = i >> 5, n = (i & 31) * 4;
      int k = 2 * p;
      const float4 f0 = b0pre[it], f1 = b1pre[it];
      v2bf d;
      d.x = (bf16_t)f0.x; d.y = (bf16_t)f1.x; *(v2bf*)(&Bs[buf][n + 0][k]) = d;
      d.x = (bf16_t)f0.y; d.y = (bf16_t)f1.y; *(v2bf*)(&Bs[buf][n + 1][k]) = d;
      d.x = (bf16_t)f0.z; d.y = (bf16_t)f1.z; *(v2bf*)(&Bs[buf][n + 2][k]) = d;
      d.x = (bf16_t)f0.w; d.y = (bf16_t)f1.w; *(v2bf*)(&Bs[buf][n + 3][k]) = d;
    }
  };

  v8f acc[4][2];
  #pragma unroll
  for (int i = 0; i < 4; ++i)
    #pragma unroll
    for (int j = 0; j < 2; ++j)
      acc[i][j] = (v8f){0.f, 0.f, 0.f, 0.f, 0.f, 0.f, 0.f, 0.f};

  stage_A_async(0, 0);
  load_B(0);
  store_B(0);
  wait_asynccnt0();
  __syncthreads();

  int buf = 0;
  for (int k0 = 0; k0 < Kd; k0 += 32, buf ^= 1) {
    const bool has_next = (k0 + 32) < Kd;
    if (has_next) {
      stage_A_async(buf ^ 1, k0 + 32);  // DMA overlaps WMMA below
      load_B(k0 + 32);
    }

    v16bf afr[4], bfr[2];
    #pragma unroll
    for (int wm = 0; wm < 4; ++wm)
      afr[wm] = load_frag_row(&As[buf][wmb + wm * 16 + lrow][0], kbase);
    #pragma unroll
    for (int wn = 0; wn < 2; ++wn)
      bfr[wn] = load_frag_row(&Bs[buf][wnb + wn * 16 + lrow][0], kbase);

    #pragma unroll
    for (int wm = 0; wm < 4; ++wm)
      #pragma unroll
      for (int wn = 0; wn < 2; ++wn)
        acc[wm][wn] = __builtin_amdgcn_wmma_f32_16x16x32_bf16(
            false, afr[wm], false, bfr[wn], (short)0, acc[wm][wn], false, false);

    if (has_next) store_B(buf ^ 1);
    wait_asynccnt0();
    __syncthreads();
  }

  // Epilogue: C/D layout => lane = col (lrow), VGPR r = row r (+8 for hi lanes)
  const int rofs = (lane < 16) ? 0 : 8;
  #pragma unroll
  for (int wm = 0; wm < 4; ++wm) {
    #pragma unroll
    for (int wn = 0; wn < 2; ++wn) {
      int col = n0 + wnb + wn * 16 + lrow;
      if (col < N) {
        float bv = bias[col];
        int mrow = m0 + wmb + wm * 16 + rofs;
        #pragma unroll
        for (int r = 0; r < 8; ++r) {
          float v = acc[wm][wn][r] + bv;
          size_t o = (size_t)(mrow + r) * N + col;
          if (OUT_BF16) ((bf16_t*)outp)[o] = (bf16_t)v;
          else          ((float*)outp)[o]  = v;
        }
      }
    }
  }
}

// ---------------------------------------------------------------------------
extern "C" void kernel_launch(void* const* d_in, const int* in_sizes, int n_in,
                              void* d_out, int out_size, void* d_ws, size_t ws_size,
                              hipStream_t stream) {
  const float* user_rep = (const float*)d_in[0];
  const int*   indices  = (const int*)d_in[1];
  const float* item_emb = (const float*)d_in[2];
  const float* W_fuse   = (const float*)d_in[3];
  const float* b_fuse   = (const float*)d_in[4];
  const float* W_proj   = (const float*)d_in[5];
  const float* b_proj   = (const float*)d_in[6];

  // workspace: A_fuse bf16 [2048][1024] (4 MiB) + fused bf16 [2048][512] (2 MiB)
  bf16_t* A_fuse = (bf16_t*)d_ws;
  bf16_t* fused  = (bf16_t*)((char*)d_ws + (size_t)B_DIM * 2 * H_DIM * sizeof(bf16_t));

  rag_prep_kernel<<<dim3(H_DIM / 256, B_DIM), 256, 0, stream>>>(
      user_rep, indices, item_emb, A_fuse);

  gemm_bf16_tn<true><<<dim3(B_DIM / 128, H_DIM / 128), 256, 0, stream>>>(
      A_fuse, W_fuse, b_fuse, (void*)fused, B_DIM, H_DIM, 2 * H_DIM);

  gemm_bf16_tn<false><<<dim3(B_DIM / 128, (V_DIM + 127) / 128), 256, 0, stream>>>(
      fused, W_proj, b_proj, d_out, B_DIM, V_DIM, H_DIM);
}
